// DeblendingTransformerBlock_18708877541727
// MI455X (gfx1250) — compile-verified, matrix operand order checked by reading, not run on hardware
//
#include <hip/hip_runtime.h>
#include <hip/hip_bf16.h>
#include <math.h>

// ---------------- CDNA5 WMMA plumbing (gfx1250, wave32) ----------------
typedef __attribute__((ext_vector_type(16))) _Float16 v16h;
typedef __attribute__((ext_vector_type(8)))  _Float16 v8h;
typedef __attribute__((ext_vector_type(8)))  float    v8f;

#define WMMA(a, b, c) __builtin_amdgcn_wmma_f32_16x16x32_f16( \
    false, (a), false, (b), (short)0, (c), false, false)

// For v_wmma_f32_16x16x32_f16: element e (0..15) of the per-lane v16h maps to
// K = ((e<8)? e : e+8) + 8*(lane>=16). A rows / B cols = lane & 15.
__device__ __forceinline__ int kmap(int e, int half) {
  return ((e < 8) ? e : e + 8) + (half << 3);
}

// f16 fragment load: p = row_base + k0. Per lane the needed halves are two
// contiguous 16-byte runs: [8*half .. +7] and [16+8*half .. +7].
__device__ __forceinline__ v16h ldfrag(const _Float16* __restrict__ p, int half) {
  const v8h lo = *(const v8h*)(p + (half << 3));
  const v8h hi = *(const v8h*)(p + 16 + (half << 3));
  v16h r;
#pragma unroll
  for (int i = 0; i < 8; ++i) { r[i] = lo[i]; r[i + 8] = hi[i]; }
  return r;
}

// ---------------- problem constants ----------------
#define Bb   2
#define Ss   64
#define NWw  1024
#define Cc   64
#define NHh  8
#define Dd   8192
#define Ff   196608          // NW*3C  (flat qkv per (b,s))
#define HIDh 256
#define HWp  65536           // H*W
#define SCALEc 4.898979485566356f   // sqrt(3*head_dim) = sqrt(24)
#define EPSc 1e-5f

// window partition: xw[b,s,nw,c] = x[b, hwidx(nw,s), c]
__device__ __forceinline__ int hwidx(int nw, int s) {
  int wh = nw >> 5, ww = nw & 31, i = s >> 3, j = s & 7;
  return ((wh << 3) + i) * 256 + (ww << 3) + j;
}

// ============ Kernel 0: convert weight matrices to f16 once ============
__global__ __launch_bounds__(256) void k_cvtw(const float* __restrict__ qkv_w,
                                              const float* __restrict__ proj_w,
                                              const float* __restrict__ fc1_w,
                                              const float* __restrict__ fc2_w,
                                              _Float16* __restrict__ wq,
                                              _Float16* __restrict__ wp,
                                              _Float16* __restrict__ w1,
                                              _Float16* __restrict__ w2) {
  int i = blockIdx.x * 256 + threadIdx.x;       // 0..16383
  if (i < 12288) wq[i] = (_Float16)qkv_w[i];
  if (i < 4096)  wp[i] = (_Float16)proj_w[i];
  w1[i] = (_Float16)fc1_w[i];
  w2[i] = (_Float16)fc2_w[i];
}

// ================= Kernel 1: QKV GEMM (131072 x 192 x 64) =================
// Yh[b][s][nw*192 + c3] = f16( sum_c xw[b,s,nw,c]*qkv_w[c3][c] + qkv_b[c3] )
__global__ __launch_bounds__(256) void k_qkv(const float* __restrict__ x,
                                             const _Float16* __restrict__ wq,
                                             const float* __restrict__ qkv_b,
                                             _Float16* __restrict__ Yh) {
  int wid  = (blockIdx.x * blockDim.x + threadIdx.x) >> 5;   // 0..8191
  int lane = threadIdx.x & 31;
  int bs = wid >> 6;                 // b*64 + s
  int mt = wid & 63;                 // 16-row tile over nw
  int b = bs >> 6, s = bs & 63;
  int m = lane & 15, half = lane >> 4;

  int nw_a = mt * 16 + m;
  const float* arow = x + ((size_t)b * HWp + hwidx(nw_a, s)) * Cc;

  v8f acc[12];
#pragma unroll
  for (int nt = 0; nt < 12; ++nt) acc[nt] = (v8f){};

#pragma unroll
  for (int ks = 0; ks < 2; ++ks) {
    v16h a;
#pragma unroll
    for (int e = 0; e < 16; ++e) a[e] = (_Float16)arow[ks * 32 + kmap(e, half)];
#pragma unroll
    for (int nt = 0; nt < 12; ++nt) {
      v16h bf = ldfrag(wq + (nt * 16 + m) * Cc + ks * 32, half);
      acc[nt] = WMMA(a, bf, acc[nt]);
    }
  }
#pragma unroll
  for (int nt = 0; nt < 12; ++nt) {
    int n = nt * 16 + m;
    float bias = qkv_b[n];
#pragma unroll
    for (int r = 0; r < 8; ++r) {
      int nw = mt * 16 + r + (half << 3);
      Yh[(size_t)bs * Ff + nw * 192 + n] = (_Float16)(acc[nt][r] + bias);
    }
  }
}

// ========== Kernel 2: attn = softmax(SCALE * Q K^T + bias), per (b,h) ==========
// Q[s,d] = Yh[b][s][h*8192+d], K[t,d] = Yh[b][t][65536 + h*8192 + d]
__global__ __launch_bounds__(512) void k_attn(const _Float16* __restrict__ Yh,
                                              const float* __restrict__ rpb,
                                              _Float16* __restrict__ attnH) {
  __shared__ float sc[64][65];
  int bh = blockIdx.x;                    // 0..15
  int b = bh >> 3, h = bh & 7;
  int tid = threadIdx.x;
  int wave = tid >> 5, lane = tid & 31;
  int mt = wave >> 2, nt = wave & 3;
  int m = lane & 15, half = lane >> 4;

  const _Float16* Yb = Yh + (size_t)b * Ss * Ff;
  const _Float16* qrow = Yb + (size_t)(mt * 16 + m) * Ff + h * Dd;
  const _Float16* krow = Yb + (size_t)(nt * 16 + m) * Ff + 65536 + h * Dd;

  v8f acc = {};
  for (int k0 = 0; k0 < Dd; k0 += 32) {
    __builtin_prefetch(qrow + k0 + 1024, 0, 1);   // global_prefetch_b8, 2KB ahead
    __builtin_prefetch(krow + k0 + 1024, 0, 1);
    v16h a  = ldfrag(qrow + k0, half);
    v16h bf = ldfrag(krow + k0, half);
    acc = WMMA(a, bf, acc);
  }
#pragma unroll
  for (int r = 0; r < 8; ++r) sc[mt * 16 + r + (half << 3)][nt * 16 + m] = acc[r];
  __syncthreads();

  if (tid < 64) {
    int row = tid, i1 = row >> 3, j1 = row & 7;
    float vals[64], vmax = -1e30f;
    for (int col = 0; col < 64; ++col) {
      int i2 = col >> 3, j2 = col & 7;
      float bias = rpb[((i1 - i2 + 7) * 15 + (j1 - j2 + 7)) * NHh + h];
      float v = sc[row][col] * SCALEc + bias;
      vals[col] = v;
      vmax = fmaxf(vmax, v);
    }
    float ssum = 0.f;
    for (int col = 0; col < 64; ++col) { vals[col] = __expf(vals[col] - vmax); ssum += vals[col]; }
    float inv = 1.0f / ssum;
    _Float16* arow = attnH + ((size_t)bh * 64 + row) * 64;
    for (int col = 0; col < 64; ++col) arow[col] = (_Float16)(vals[col] * inv);
  }
}

// ========== Kernel 3a: Z = attn @ V  (64 x 8192 x 64 per (b,h)) ==========
// V[t,d] = Yh[b][t][131072 + h*8192 + d]; Z overwrites Yh's dead Q region:
// Yh[b][s][h*8192 + d]  -> later read as Z[b][s][nw*64 + c]
__global__ __launch_bounds__(256) void k_av(_Float16* __restrict__ Yh,
                                            const _Float16* __restrict__ attnH) {
  int wid  = (blockIdx.x * blockDim.x + threadIdx.x) >> 5;   // 0..4095
  int lane = threadIdx.x & 31;
  int bh = wid >> 8;
  int rem = wid & 255;
  int mt = rem >> 6, ng = rem & 63;        // 8 n-tiles per group
  int b = bh >> 3, h = bh & 7;
  int m = lane & 15, half = lane >> 4;

  const _Float16* arow = attnH + ((size_t)bh * 64 + mt * 16 + m) * 64;
  v16h afrag[2];
#pragma unroll
  for (int ks = 0; ks < 2; ++ks) afrag[ks] = ldfrag(arow + ks * 32, half);

  const _Float16* Yb = Yh + (size_t)b * Ss * Ff;
#pragma unroll
  for (int g = 0; g < 8; ++g) {
    int nt = ng * 8 + g;
    int d0 = nt * 16 + m;
    v8f acc = {};
#pragma unroll
    for (int ks = 0; ks < 2; ++ks) {
      v16h bf;
#pragma unroll
      for (int e = 0; e < 16; ++e) {
        int t = ks * 32 + kmap(e, half);
        bf[e] = Yb[(size_t)t * Ff + 131072 + h * Dd + d0];
      }
      acc = WMMA(afrag[ks], bf, acc);
    }
#pragma unroll
    for (int r = 0; r < 8; ++r) {
      int srow = mt * 16 + r + (half << 3);
      Yh[((size_t)b * Ss + srow) * Ff + h * Dd + d0] = (_Float16)acc[r];
    }
  }
}

// ===== Kernel 3b: proj + window_reverse + LayerNorm + residual -> x1 =====
__global__ __launch_bounds__(256) void k_proj(const _Float16* __restrict__ Yh,
                                              const _Float16* __restrict__ wp,
                                              const float* __restrict__ proj_b,
                                              const float* __restrict__ x,
                                              const float* __restrict__ g1,
                                              const float* __restrict__ b1,
                                              float* __restrict__ x1) {
  int wid  = (blockIdx.x * blockDim.x + threadIdx.x) >> 5;   // 0..8191
  int lane = threadIdx.x & 31;
  int bs = wid >> 6, nwt = wid & 63;
  int b = bs >> 6, s = bs & 63;
  int m = lane & 15, half = lane >> 4;

  const _Float16* zrow = Yh + (size_t)bs * Ff + (nwt * 16 + m) * 64;
  v16h afrag[2];
#pragma unroll
  for (int ks = 0; ks < 2; ++ks) afrag[ks] = ldfrag(zrow + ks * 32, half);

  v8f acc[4];
#pragma unroll
  for (int nt = 0; nt < 4; ++nt) {
    v8f a = {};
#pragma unroll
    for (int ks = 0; ks < 2; ++ks) {
      v16h bf = ldfrag(wp + (nt * 16 + m) * Cc + ks * 32, half);
      a = WMMA(afrag[ks], bf, a);
    }
    acc[nt] = a;
  }

#pragma unroll
  for (int r = 0; r < 8; ++r) {
    float v[4], p = 0.f, q = 0.f;
#pragma unroll
    for (int nt = 0; nt < 4; ++nt) {
      v[nt] = acc[nt][r] + proj_b[nt * 16 + m];
      p += v[nt];
      q += v[nt] * v[nt];
    }
#pragma unroll
    for (int off = 1; off < 16; off <<= 1) {       // reduce within 16-lane half
      p += __shfl_xor(p, off, 32);
      q += __shfl_xor(q, off, 32);
    }
    float mean = p * (1.0f / 64.0f);
    float var  = q * (1.0f / 64.0f) - mean * mean;
    float rstd = rsqrtf(var + EPSc);
    int nw = nwt * 16 + r + (half << 3);
    size_t base = ((size_t)b * HWp + hwidx(nw, s)) * Cc;
#pragma unroll
    for (int nt = 0; nt < 4; ++nt) {
      int c = nt * 16 + m;
      x1[base + c] = x[base + c] + (v[nt] - mean) * rstd * g1[c] + b1[c];
    }
  }
}

// ===== Kernel 4: fused MLP: out = x1 + LN( fc2( gelu( fc1(x1) ) ) ) =====
__global__ __launch_bounds__(128) void k_mlp(const float* __restrict__ x1,
                                             const _Float16* __restrict__ w1,
                                             const float* __restrict__ fc1_b,
                                             const _Float16* __restrict__ w2,
                                             const float* __restrict__ fc2_b,
                                             const float* __restrict__ g2,
                                             const float* __restrict__ b2,
                                             float* __restrict__ out) {
  __shared__ __align__(16) _Float16 hbuf[4][16][HIDh];   // 32KB: 16x256 f16 per wave
  int wv = threadIdx.x >> 5, lane = threadIdx.x & 31;
  int tile = blockIdx.x * 4 + wv;                        // 0..8191
  int m = lane & 15, half = lane >> 4;
  size_t row0 = (size_t)tile * 16;

  const float* arow = x1 + (row0 + m) * Cc;
  v16h afrag[2];
#pragma unroll
  for (int ks = 0; ks < 2; ++ks)
#pragma unroll
    for (int e = 0; e < 16; ++e) afrag[ks][e] = (_Float16)arow[ks * 32 + kmap(e, half)];

  // fc1 + exact GELU, one 16-col tile at a time
#pragma unroll 2
  for (int nt = 0; nt < 16; ++nt) {
    v8f acc = {};
#pragma unroll
    for (int ks = 0; ks < 2; ++ks) {
      v16h bf = ldfrag(w1 + (nt * 16 + m) * Cc + ks * 32, half);
      acc = WMMA(afrag[ks], bf, acc);
    }
    float bias = fc1_b[nt * 16 + m];
#pragma unroll
    for (int r = 0; r < 8; ++r) {
      float v = acc[r] + bias;
      v = 0.5f * v * (1.0f + erff(v * 0.70710678118f));
      hbuf[wv][r + (half << 3)][nt * 16 + m] = (_Float16)v;
    }
  }
  __syncthreads();

  // fc2: K = 256 from LDS (ds_load_b128 A-fragments)
  v8f acc2[4];
#pragma unroll
  for (int nt = 0; nt < 4; ++nt) {
    v8f acc = {};
#pragma unroll
    for (int ks = 0; ks < 8; ++ks) {
      v16h af = ldfrag(&hbuf[wv][m][0] + ks * 32, half);
      v16h bf = ldfrag(w2 + (nt * 16 + m) * HIDh + ks * 32, half);
      acc = WMMA(af, bf, acc);
    }
    acc2[nt] = acc;
  }

  // LayerNorm + residual
#pragma unroll
  for (int r = 0; r < 8; ++r) {
    float v[4], p = 0.f, q = 0.f;
#pragma unroll
    for (int nt = 0; nt < 4; ++nt) {
      v[nt] = acc2[nt][r] + fc2_b[nt * 16 + m];
      p += v[nt];
      q += v[nt] * v[nt];
    }
#pragma unroll
    for (int off = 1; off < 16; off <<= 1) {
      p += __shfl_xor(p, off, 32);
      q += __shfl_xor(q, off, 32);
    }
    float mean = p * (1.0f / 64.0f);
    float var  = q * (1.0f / 64.0f) - mean * mean;
    float rstd = rsqrtf(var + EPSc);
    size_t row = row0 + r + (half << 3);
#pragma unroll
    for (int nt = 0; nt < 4; ++nt) {
      int c = nt * 16 + m;
      out[row * Cc + c] = x1[row * Cc + c] + (v[nt] - mean) * rstd * g2[c] + b2[c];
    }
  }
}

// ================= launcher =================
extern "C" void kernel_launch(void* const* d_in, const int* in_sizes, int n_in,
                              void* d_out, int out_size, void* d_ws, size_t ws_size,
                              hipStream_t stream) {
  const float* x      = (const float*)d_in[0];
  const float* qkv_w  = (const float*)d_in[1];
  const float* qkv_b  = (const float*)d_in[2];
  const float* proj_w = (const float*)d_in[3];
  const float* proj_b = (const float*)d_in[4];
  const float* rpb    = (const float*)d_in[5];
  const float* n1g    = (const float*)d_in[6];
  const float* n1b    = (const float*)d_in[7];
  const float* n2g    = (const float*)d_in[8];
  const float* n2b    = (const float*)d_in[9];
  const float* fc1_w  = (const float*)d_in[10];
  const float* fc1_b  = (const float*)d_in[11];
  const float* fc2_w  = (const float*)d_in[12];
  const float* fc2_b  = (const float*)d_in[13];
  float* out = (float*)d_out;

  // workspace (f16-heavy): Yh 50.3MB (attn@V overwrites dead Q third),
  // attnH 128KB, f16 weights ~96KB, x1 f32 33.6MB -> ~84MB total
  _Float16* Yh    = (_Float16*)d_ws;                       // 25165824 halves
  _Float16* attnH = Yh + (size_t)Bb * Ss * Ff;             // 65536
  _Float16* wq    = attnH + (size_t)Bb * NHh * Ss * Ss;    // 12288
  _Float16* wp    = wq + 12288;                            // 4096
  _Float16* w1    = wp + 4096;                             // 16384
  _Float16* w2    = w1 + 16384;                            // 16384
  float*    x1    = (float*)(w2 + 16384);                  // 8388608 f32

  k_cvtw<<<  64, 256, 0, stream>>>(qkv_w, proj_w, fc1_w, fc2_w, wq, wp, w1, w2);
  k_qkv <<<1024, 256, 0, stream>>>(x, wq, qkv_b, Yh);
  k_attn<<<  16, 512, 0, stream>>>(Yh, rpb, attnH);
  k_av  <<< 512, 256, 0, stream>>>(Yh, attnH);
  k_proj<<<1024, 256, 0, stream>>>(Yh, wp, proj_b, x, n1g, n1b, x1);
  k_mlp <<<2048, 128, 0, stream>>>(x1, w1, fc1_b, w2, fc2_b, n2g, n2b, out);
}
